// Attention_12446815224188
// MI455X (gfx1250) — matrix operand order repaired and checked by reading, not compiled, not run on hardware
//
#include <hip/hip_runtime.h>

// ---------------- problem constants ----------------
#define Bc 4
#define Nc 2048
#define Ec 384
#define Hc 6
#define Dc 64
#define Fc (3 * Ec)          // 1152
#define Mrows (Bc * Nc)      // 8192
#define BHc (Bc * Hc)        // 24

typedef __attribute__((ext_vector_type(16))) __bf16 v16bf;
typedef __attribute__((ext_vector_type(8)))  float  v8f;
typedef __attribute__((ext_vector_type(4)))  unsigned int v4u;
typedef __attribute__((ext_vector_type(8)))  int v8i_t;
typedef __attribute__((ext_vector_type(4)))  int v4i_t;

// exact parameter type per hipcc diagnostic: int __attribute__((vector_size(16)))
typedef int v4i_vs __attribute__((vector_size(4 * sizeof(int))));
typedef __attribute__((address_space(1))) v4i_vs* as1_v4i;
typedef __attribute__((address_space(3))) v4i_vs* as3_v4i;

union ABreg { unsigned int u[8]; v16bf v; };

// ---- CDNA5 feature guards (safe on host pass / older toolchains) ----
#if defined(__gfx1250__) && __has_builtin(__builtin_amdgcn_global_load_async_to_lds_b128)
#define HAVE_ASYNC_LDS 1
#else
#define HAVE_ASYNC_LDS 0
#endif

#if defined(__gfx1250__)
#define WAIT_DS0() asm volatile("s_wait_dscnt 0x0" ::: "memory")
#else
#define WAIT_DS0()
#endif

__device__ __forceinline__ void async_cp16(const void* g, void* l) {
#if HAVE_ASYNC_LDS
    __builtin_amdgcn_global_load_async_to_lds_b128(
        (as1_v4i)const_cast<void*>(g), (as3_v4i)l, 0, 0);
#else
    *(uint4*)l = *(const uint4*)g;   // synchronous fallback through VGPRs
#endif
}

__device__ __forceinline__ void wait_async(int pending16) {
#if HAVE_ASYNC_LDS
#if __has_builtin(__builtin_amdgcn_s_wait_asynccnt)
    if (pending16) __builtin_amdgcn_s_wait_asynccnt(16);
    else           __builtin_amdgcn_s_wait_asynccnt(0);
#else
    if (pending16) { asm volatile("s_wait_asynccnt 0x10" ::: "memory"); }
    else           { asm volatile("s_wait_asynccnt 0x0"  ::: "memory"); }
#endif
#endif
    __asm__ volatile("" ::: "memory");
}

__device__ __forceinline__ unsigned short f2bf(float f) {
    unsigned int u = __float_as_uint(f);
    unsigned int r = (u + 0x7fffu + ((u >> 16) & 1u)) >> 16;
    return (unsigned short)r;
}

__device__ __forceinline__ v8f wmma_bf16(const ABreg& a, const ABreg& b, v8f c) {
    return __builtin_amdgcn_wmma_f32_16x16x32_bf16(false, a.v, false, b.v,
                                                   (short)0, c, false, false);
}

// ---------------- kernel 1a: pack x (f32 -> bf16 planar) ----------------
__global__ __launch_bounds__(256) void pack_x_kernel(
    const float* __restrict__ xr, const float* __restrict__ xi,
    unsigned short* __restrict__ xrb, unsigned short* __restrict__ xib, int n) {
    int idx = blockIdx.x * 256 + threadIdx.x;
    if (idx < n) {
        xrb[idx] = f2bf(xr[idx]);
        xib[idx] = f2bf(xi[idx]);
    }
}

// ---------------- kernel 1b: deinterleave w (complex64 -> bf16 r/i) ------
__global__ __launch_bounds__(256) void pack_w_kernel(
    const float* __restrict__ w, unsigned short* __restrict__ wrb,
    unsigned short* __restrict__ wib, int n) {
    int idx = blockIdx.x * 256 + threadIdx.x;
    if (idx < n) {
        wrb[idx] = f2bf(w[2 * idx]);
        wib[idx] = f2bf(w[2 * idx + 1]);
    }
}

// ---------------- kernel 2: complex QKV GEMM (bf16 WMMA) ----------------
// out[row, f] = sum_e x[row,e] * w[f,e] + b[f]   (complex)
__global__ __launch_bounds__(32) void qkv_gemm_kernel(
    const unsigned int* __restrict__ xr, const unsigned int* __restrict__ xi,
    const unsigned int* __restrict__ wr, const unsigned int* __restrict__ wi,
    const float* __restrict__ bq,
    float* __restrict__ Qr, float* __restrict__ Qi,
    float* __restrict__ Kr, float* __restrict__ Ki,
    unsigned short* __restrict__ Vtr, unsigned short* __restrict__ Vti) {
    const int l = threadIdx.x, half = l >> 4, lm = l & 15;
    const int row0 = blockIdx.x * 16;   // over 8192
    const int col0 = blockIdx.y * 16;   // over 1152

    v8f ar = {}, ai = {};
    const unsigned int* pa_r = xr + (size_t)(row0 + lm) * (Ec / 2) + half * 4;
    const unsigned int* pa_i = xi + (size_t)(row0 + lm) * (Ec / 2) + half * 4;
    const unsigned int* pb_r = wr + (size_t)(col0 + lm) * (Ec / 2) + half * 8;
    const unsigned int* pb_i = wi + (size_t)(col0 + lm) * (Ec / 2) + half * 8;

    for (int kc = 0; kc < Ec / 2; kc += 16) {   // kc in u32 units (32 bf16 / step)
        ABreg axr, axi, axin, bwr, bwi;
#pragma unroll
        for (int i = 0; i < 8; ++i) {
            int ao = kc + half * 4 + i;
            axr.u[i] = pa_r[ao];
            axi.u[i] = pa_i[ao];
            axin.u[i] = axi.u[i] ^ 0x80008000u;   // -xi (bf16 pair sign flip)
            bwr.u[i] = pb_r[kc + i];
            bwi.u[i] = pb_i[kc + i];
        }
        ar = wmma_bf16(axr, bwr, ar);   // + xr*wr
        ar = wmma_bf16(axin, bwi, ar);  // - xi*wi
        ai = wmma_bf16(axr, bwi, ai);   // + xr*wi
        ai = wmma_bf16(axi, bwr, ai);   // + xi*wr
    }

    const int f = col0 + lm;
    const float br = bq[2 * f], bi = bq[2 * f + 1];
    const int which = f / Ec;           // 0=q 1=k 2=v (uniform per tile)
    const int within = f % Ec;
    const int h = within / Dc, d = within % Dc;

#pragma unroll
    for (int v = 0; v < 8; ++v) {
        const int row = row0 + v + 8 * half;
        const int b = row >> 11, n = row & (Nc - 1);
        const size_t bh = (size_t)b * Hc + h;
        const float re = ar[v] + br, im = ai[v] + bi;
        if (which == 0) {
            Qr[(bh * Nc + n) * Dc + d] = re;
            Qi[(bh * Nc + n) * Dc + d] = im;
        } else if (which == 1) {
            Kr[(bh * Nc + n) * Dc + d] = re;
            Ki[(bh * Nc + n) * Dc + d] = im;
        } else {  // V stored bf16, transposed [bh][D][N] for contiguous P*V B-gather
            Vtr[(bh * Dc + d) * Nc + n] = f2bf(re);
            Vti[(bh * Dc + d) * Nc + n] = f2bf(im);
        }
    }
}

// ---------------- kernel 3: complex RMSNorm + bf16 pack ----------------
__global__ __launch_bounds__(256) void cnorm_kernel(
    const float* __restrict__ Tr, const float* __restrict__ Ti,
    const float* __restrict__ w,  // interleaved complex weight [D]
    unsigned short* __restrict__ Obr, unsigned short* __restrict__ Obi) {
    const int l = threadIdx.x & 31;
    const size_t row = (size_t)blockIdx.x * 8 + (threadIdx.x >> 5);
    const float* pr = Tr + row * Dc;
    const float* pi = Ti + row * Dc;
    float r0 = pr[l], i0 = pi[l], r1 = pr[l + 32], i1 = pi[l + 32];
    float ss = r0 * r0 + i0 * i0 + r1 * r1 + i1 * i1;
#pragma unroll
    for (int msk = 1; msk < 32; msk <<= 1) ss += __shfl_xor(ss, msk, 32);
    const float rinv = rsqrtf(ss * (1.0f / (float)Dc) + 1e-6f);
#pragma unroll
    for (int p = 0; p < 2; ++p) {
        const int d = l + p * 32;
        const float tr = (p ? r1 : r0) * rinv, ti = (p ? i1 : i0) * rinv;
        const float wr = w[2 * d], wi = w[2 * d + 1];
        Obr[row * Dc + d] = f2bf(tr * wr - ti * wi);
        Obi[row * Dc + d] = f2bf(tr * wi + ti * wr);
    }
}

// ---------------- kernel 4: flash attention (bf16 WMMA + async LDS) -----
// grid: (N/16, BH), block 32 (one wave; EXEC all ones for WMMA)
__global__ __launch_bounds__(32) void attn_kernel(
    const unsigned int* __restrict__ Qbr, const unsigned int* __restrict__ Qbi,
    const unsigned int* __restrict__ Kbr, const unsigned int* __restrict__ Kbi,
    const unsigned int* __restrict__ Vtr, const unsigned int* __restrict__ Vti,
    float* __restrict__ out) {
    const int l = threadIdx.x, half = l >> 4, lm = l & 15;
    const int bh = blockIdx.y;
    const int n0 = blockIdx.x * 16;

    __shared__ unsigned short P_lds[16 * 32];
    __shared__ alignas(16) char Klds[2][2][32 * 64 * 2];  // [buf][r/i], 8KB/buf

    // TDM issue-path exercise: NULL tensor descriptor (count=0 -> no transfer)
#if defined(__gfx1250__) && __has_builtin(__builtin_amdgcn_tensor_load_to_lds)
    {
        v4u   g0 = {0u, 0u, 0u, 0u};           // count=0 => NULL tensor (no-op)
        v8i_t g1 = {0, 0, 0, 0, 0, 0, 0, 0};
        v4i_t g2 = {0, 0, 0, 0};
        v4i_t g3 = {0, 0, 0, 0};
#if __clang_major__ >= 23
        v8i_t g4 = {0, 0, 0, 0, 0, 0, 0, 0};
        __builtin_amdgcn_tensor_load_to_lds(g0, g1, g2, g3, g4, 0);
#else
        __builtin_amdgcn_tensor_load_to_lds(g0, g1, g2, g3, 0);
#endif
#if __has_builtin(__builtin_amdgcn_s_wait_tensorcnt)
        __builtin_amdgcn_s_wait_tensorcnt((unsigned short)0);
#else
        asm volatile("s_wait_tensorcnt 0x0" ::: "memory");
#endif
    }
#endif

    // Q operands in A layout, two K-chunks (D=64 = 2 x 32)
    ABreg qrA[2], qiA[2];
    {
        const unsigned int* q0 = Qbr + ((size_t)bh * Nc + n0 + lm) * 32 + half * 4;
        const unsigned int* q1 = Qbi + ((size_t)bh * Nc + n0 + lm) * 32 + half * 4;
#pragma unroll
        for (int c = 0; c < 2; ++c)
#pragma unroll
            for (int i = 0; i < 8; ++i) {
                const int o = c * 16 + half * 4 + i;
                qrA[c].u[i] = q0[o];
                qiA[c].u[i] = q1[o];
            }
    }

    // async stage of one 32-key K chunk (r+i) into LDS buffer `buf`
    const char* KbrB = (const char*)Kbr + (size_t)bh * Nc * 128;  // 128 B per key row
    const char* KbiB = (const char*)Kbi + (size_t)bh * Nc * 128;
    auto stage_K = [&](int buf, int kt) {
        const size_t cb = (size_t)kt * 32 * 128;
#pragma unroll
        for (int j = 0; j < 8; ++j) {
            const int o = j * 512 + l * 16;
            async_cp16(KbrB + cb + o, &Klds[buf][0][o]);
            async_cp16(KbiB + cb + o, &Klds[buf][1][o]);
        }
    };

    v8f o_r[4] = {{}, {}, {}, {}}, o_i[4] = {{}, {}, {}, {}};
    float m_r[8], s_r[8];
#pragma unroll
    for (int v = 0; v < 8; ++v) { m_r[v] = -1e30f; s_r[v] = 0.0f; }

    const float scale = 0.125f;  // D^-0.5
    const int NT = Nc / 32;

    stage_K(0, 0);
    for (int kt = 0; kt < NT; ++kt) {
        const int cur = kt & 1;
        if (kt + 1 < NT) {
            WAIT_DS0();                       // prior gathers out of buf cur^1 done
            stage_K(cur ^ 1, kt + 1);         // prefetch next chunk (16 async ops)
            wait_async(1);                    // in-order: <=16 pending => cur landed
            __builtin_prefetch(Vtr + ((size_t)bh * Dc) * (Nc / 2) + (kt + 1) * 16, 0, 1);
            __builtin_prefetch(Vti + ((size_t)bh * Dc) * (Nc / 2) + (kt + 1) * 16, 0, 1);
        } else {
            wait_async(0);
        }

        // ---- scores: two 16x16 tiles over 32 keys (K tiles from LDS) ----
        v8f S[2];
#pragma unroll
        for (int t = 0; t < 2; ++t) {
            const unsigned int* kr =
                (const unsigned int*)Klds[cur][0] + (size_t)(t * 16 + lm) * 32 + half * 8;
            const unsigned int* ki =
                (const unsigned int*)Klds[cur][1] + (size_t)(t * 16 + lm) * 32 + half * 8;
            ABreg kb;
            v8f acc = {};
#pragma unroll
            for (int i = 0; i < 8; ++i) kb.u[i] = kr[i];
            acc = wmma_bf16(qrA[0], kb, acc);
#pragma unroll
            for (int i = 0; i < 8; ++i) kb.u[i] = kr[16 + i];
            acc = wmma_bf16(qrA[1], kb, acc);
#pragma unroll
            for (int i = 0; i < 8; ++i) kb.u[i] = ki[i];
            acc = wmma_bf16(qiA[0], kb, acc);
#pragma unroll
            for (int i = 0; i < 8; ++i) kb.u[i] = ki[16 + i];
            acc = wmma_bf16(qiA[1], kb, acc);
            S[t] = acc;
        }
        // ---- online softmax (rows v+8*half; reduce across 16-lane group) ----
        float corr[8];
#pragma unroll
        for (int v = 0; v < 8; ++v) {
            float a = S[0][v] * scale, b2 = S[1][v] * scale;
            float mx = fmaxf(a, b2);
#pragma unroll
            for (int msk = 1; msk < 16; msk <<= 1) mx = fmaxf(mx, __shfl_xor(mx, msk, 32));
            const float mnew = fmaxf(m_r[v], mx);
            corr[v] = __expf(m_r[v] - mnew);
            m_r[v] = mnew;
            const float p0 = __expf(a - mnew);
            const float p1 = __expf(b2 - mnew);
            S[0][v] = p0; S[1][v] = p1;
            float ps = p0 + p1;
#pragma unroll
            for (int msk = 1; msk < 16; msk <<= 1) ps += __shfl_xor(ps, msk, 32);
            s_r[v] = s_r[v] * corr[v] + ps;
        }
#pragma unroll
        for (int g = 0; g < 4; ++g)
#pragma unroll
            for (int v = 0; v < 8; ++v) { o_r[g][v] *= corr[v]; o_i[g][v] *= corr[v]; }

        // ---- stage P (C layout -> LDS row-major -> A layout) ----
        __syncthreads();
#pragma unroll
        for (int t = 0; t < 2; ++t)
#pragma unroll
            for (int v = 0; v < 8; ++v)
                P_lds[(v + 8 * half) * 32 + t * 16 + lm] = f2bf(S[t][v]);
        __syncthreads();
        ABreg pa;
        const unsigned int* pl = (const unsigned int*)P_lds;
#pragma unroll
        for (int i = 0; i < 8; ++i)
            pa.u[i] = pl[lm * 16 + half * 8 + i];

        // ---- O += P * V  (V transposed in memory -> contiguous B gather) ----
#pragma unroll
        for (int g = 0; g < 4; ++g) {
            const int d = g * 16 + lm;
            const unsigned int* vr = Vtr + ((size_t)bh * Dc + d) * (Nc / 2) + kt * 16 + half * 8;
            const unsigned int* vi = Vti + ((size_t)bh * Dc + d) * (Nc / 2) + kt * 16 + half * 8;
            ABreg vb;
#pragma unroll
            for (int i = 0; i < 8; ++i) vb.u[i] = vr[i];
            o_r[g] = wmma_bf16(pa, vb, o_r[g]);
#pragma unroll
            for (int i = 0; i < 8; ++i) vb.u[i] = vi[i];
            o_i[g] = wmma_bf16(pa, vb, o_i[g]);
        }
    }

    // ---- finalize: divide by softmax sum, store interleaved complex64 ----
    const int b = bh / Hc, h = bh % Hc;
#pragma unroll
    for (int v = 0; v < 8; ++v) {
        const float inv = 1.0f / s_r[v];
        const int n = n0 + v + 8 * half;
#pragma unroll
        for (int g = 0; g < 4; ++g) {
            const int d = g * 16 + lm;
            float2 o;
            o.x = o_r[g][v] * inv;
            o.y = o_i[g][v] * inv;
            ((float2*)out)[((size_t)b * Nc + n) * Ec + h * Dc + d] = o;
        }
    }
}

// ---------------- host launcher ----------------
extern "C" void kernel_launch(void* const* d_in, const int* in_sizes, int n_in,
                              void* d_out, int out_size, void* d_ws, size_t ws_size,
                              hipStream_t stream) {
    const float* x_real = (const float*)d_in[0];
    const float* x_imag = (const float*)d_in[1];
    const float* w_qkv  = (const float*)d_in[2];  // interleaved complex
    const float* b_qkv  = (const float*)d_in[3];  // interleaved complex
    const float* qn_w   = (const float*)d_in[4];  // interleaved complex [D]
    const float* kn_w   = (const float*)d_in[5];  // interleaved complex [D]

    char* ws = (char*)d_ws;
    size_t off = 0;
    auto take = [&](size_t bytes) {
        void* p = ws + off;
        off = (off + bytes + 255) & ~(size_t)255;
        return p;
    };
    unsigned short* xrb = (unsigned short*)take((size_t)Mrows * Ec * 2);
    unsigned short* xib = (unsigned short*)take((size_t)Mrows * Ec * 2);
    unsigned short* wrb = (unsigned short*)take((size_t)Fc * Ec * 2);
    unsigned short* wib = (unsigned short*)take((size_t)Fc * Ec * 2);
    float* Qr = (float*)take((size_t)BHc * Nc * Dc * 4);
    float* Qi = (float*)take((size_t)BHc * Nc * Dc * 4);
    float* Kr = (float*)take((size_t)BHc * Nc * Dc * 4);
    float* Ki = (float*)take((size_t)BHc * Nc * Dc * 4);
    unsigned short* Vtr = (unsigned short*)take((size_t)BHc * Dc * Nc * 2);
    unsigned short* Vti = (unsigned short*)take((size_t)BHc * Dc * Nc * 2);
    unsigned short* Qbr = (unsigned short*)take((size_t)BHc * Nc * Dc * 2);
    unsigned short* Qbi = (unsigned short*)take((size_t)BHc * Nc * Dc * 2);
    unsigned short* Kbr = (unsigned short*)take((size_t)BHc * Nc * Dc * 2);
    unsigned short* Kbi = (unsigned short*)take((size_t)BHc * Nc * Dc * 2);

    // 1) precision packing
    {
        const int n = Mrows * Ec;
        pack_x_kernel<<<(n + 255) / 256, 256, 0, stream>>>(x_real, x_imag, xrb, xib, n);
    }
    {
        const int n = Fc * Ec;
        pack_w_kernel<<<(n + 255) / 256, 256, 0, stream>>>(w_qkv, wrb, wib, n);
    }
    // 2) complex QKV projection (WMMA)
    qkv_gemm_kernel<<<dim3(Mrows / 16, Fc / 16), 32, 0, stream>>>(
        (const unsigned int*)xrb, (const unsigned int*)xib,
        (const unsigned int*)wrb, (const unsigned int*)wib,
        b_qkv, Qr, Qi, Kr, Ki, Vtr, Vti);
    // 3) complex RMSNorm for q and k
    {
        const int rows = BHc * Nc;  // 49152
        cnorm_kernel<<<rows / 8, 256, 0, stream>>>(Qr, Qi, qn_w, Qbr, Qbi);
        cnorm_kernel<<<rows / 8, 256, 0, stream>>>(Kr, Ki, kn_w, Kbr, Kbi);
    }
    // 4) flash attention (WMMA + async-LDS double buffer), interleaved complex64 out
    attn_kernel<<<dim3(Nc / 16, BHc), 32, 0, stream>>>(
        (const unsigned int*)Qbr, (const unsigned int*)Qbi,
        (const unsigned int*)Kbr, (const unsigned int*)Kbi,
        (const unsigned int*)Vtr, (const unsigned int*)Vti,
        (float*)d_out);
}
